// DeepLSTM_11304353923324
// MI455X (gfx1250) — matrix operand order repaired and checked by reading, not compiled, not run on hardware
//
#include <hip/hip_runtime.h>
#include <hip/hip_bf16.h>
#include <stddef.h>

// ---------------- CDNA5 WMMA types ----------------
typedef __attribute__((ext_vector_type(16))) __bf16 v16bf;
typedef __attribute__((ext_vector_type(8)))  float  v8f;

#define SQ   512   // sequence length
#define BB   32    // batch
#define HH   512   // hidden
#define NWG  32    // workgroups in recurrent kernel (each owns 16 columns of every group)
#define NGRP 6     // i, f, g, o, r, lin
// packed weight geometry: per layer, tiles[g(6)][kt(32)][nt(32)] of 32Kx16N bf16, 512 elems each
#define TILE_E     512
#define KT_STRIDE  (32 * TILE_E)          // 16384 elems: nt-major inside a (g,kt)
#define G_STRIDE   (32 * KT_STRIDE)       // 524288 elems
#define LAYER_E    (NGRP * G_STRIDE)      // 3145728 elems per layer
#define NPACK      (2 * LAYER_E)          // total packed bf16 elements
// per-WG resident weight slice: groups 0..4 have 32 k-tiles, lin (g=5) has 16 -> 176 tiles
#define WSLOTS     176
#define SA_E       (2 * 32 * TILE_E)      // 32768 bf16 = 64 KB A-tiles
#define SW_E       (WSLOTS * TILE_E)      // 90112 bf16 = 176 KB weight tiles
#define SMEM_BYTES ((SA_E + SW_E) * 2)    // 245760 B = 240 KB dynamic LDS (WGP has 320 KB)

static __device__ inline __bf16 f2bf(float f) {
  unsigned u = __builtin_bit_cast(unsigned, f);
  unsigned r = (u + 0x7FFFu + ((u >> 16) & 1u)) >> 16;   // round-to-nearest-even
  unsigned short s = (unsigned short)r;
  return __builtin_bit_cast(__bf16, s);
}
static __device__ inline float sigf(float x)  { return 1.0f / (1.0f + __expf(-x)); }
static __device__ inline float tanhf_(float x){ float e = __expf(2.0f * x); return (e - 1.0f) / (e + 1.0f); }

#define WMMA_BF16(A, B, C) \
  __builtin_amdgcn_wmma_f32_16x16x32_bf16(false, (A), false, (B), (short)0, (C), false, false)

// ---------------- weight packing: f32 -> bf16 B-tiles, column-group interleaved ----------------
// B-tile layout (32K x 16N): element (ln, j): k = kt*32 + 16*(ln>>4) + j, n = nt*16 + (ln&15)
// Wbig rows: k<512 -> x-side weights, k>=512 -> h-side weights. Groups: 0..3 = i/f/g/o, 4 = r, 5 = lin (x only).
__global__ void pack_w_kernel(const float* __restrict__ Wih, const float* __restrict__ Whh,
                              const float* __restrict__ Wgi, const float* __restrict__ Wgs,
                              const float* __restrict__ Wli, __bf16* __restrict__ wpack) {
  int idx = blockIdx.x * 256 + threadIdx.x;
  if (idx >= NPACK) return;
  int j  = idx & 15;
  int t2 = idx >> 4;
  int ln = t2 & 31; t2 >>= 5;
  int nt = t2 & 31; t2 >>= 5;
  int kt = t2 & 31; t2 >>= 5;
  int g  = t2 % 6;
  int l  = t2 / 6;
  int k  = kt * 32 + 16 * (ln >> 4) + j;
  int n  = nt * 16 + (ln & 15);
  float v;
  if (k < HH) {                       // x-side
    if (g < 4)       v = Wih[((size_t)l * 4 * HH + (size_t)g * HH + n) * HH + k];
    else if (g == 4) v = Wgi[((size_t)l * HH + n) * HH + k];
    else             v = Wli[((size_t)l * HH + n) * HH + k];
  } else {                            // h-side
    int kh = k - HH;
    if (g < 4)       v = Whh[((size_t)l * 4 * HH + (size_t)g * HH + n) * HH + kh];
    else if (g == 4) v = Wgs[(size_t)n * HH + kh];
    else             v = 0.0f;        // lin group has no h contribution (never multiplied anyway)
  }
  wpack[idx] = f2bf(v);
}

// ---------------- per-layer init: zero barrier + h ping-pong ----------------
__global__ void init_kernel(unsigned* __restrict__ sync, float* __restrict__ hbuf) {
  int i = blockIdx.x * 256 + threadIdx.x;
  if (i < 2) sync[i] = 0u;
  if (i < 2 * BB * HH) hbuf[i] = 0.0f;
}

// ---------------- persistent recurrent kernel ----------------
// 32 WGs x 64 threads (2 waves). WG w owns columns n = 16w..16w+15 of every group.
// Weight slice (176 KB) is preloaded into LDS once; both waves feed WMMA A and B from LDS.
// Wave m computes M-tile m (batches 16m..16m+15). 176 v_wmma_f32_16x16x32_bf16 per wave per step.
// All operand loads for a K-tile are hoisted into distinct vectors so ds_load latency
// overlaps WMMA issue (partial s_wait_dscnt instead of wait-0 per WMMA).
__global__ void __launch_bounds__(64) recur_kernel(
    const float* __restrict__ xin,   const float* __restrict__ mask,
    const float* __restrict__ bih,   const float* __restrict__ bhh,
    const float* __restrict__ bgi,   const float* __restrict__ bgs,
    const float* __restrict__ dmask, const int*   __restrict__ train,
    const __bf16* __restrict__ wpack, float* __restrict__ hbuf,
    unsigned* __restrict__ sync, float* __restrict__ outp, int layer) {

  extern __shared__ __bf16 smem[];
  __bf16* sA = smem;            // [2 Mtiles][32 Ktiles][32 ln][16]  = 64 KB
  __bf16* sW = smem + SA_E;     // [176 slots][32 ln][16]            = 176 KB

  const int wave = threadIdx.x >> 5;      // 0..1  -> M-tile
  const int ln   = threadIdx.x & 31;      // lane
  const int nl   = ln & 15;
  const int n    = blockIdx.x * 16 + nl;  // owned column
  const int mrow = 8 * (ln >> 4);         // C-layout M offset for this lane

  // ---- one-time preload of this WG's weight slice into LDS (16 B chunks) ----
  // slot s: s<160 -> (g = s/32, kt = s%32); s>=160 -> (g = 5, kt = s-160)
  {
    const __bf16* wgsrc = wpack + (size_t)blockIdx.x * TILE_E;
    for (int c = threadIdx.x; c < (SW_E / 8); c += 64) {
      int s  = c >> 6;                    // tile slot (512 elems = 64 chunks)
      int cc = c & 63;
      int g  = (s < 160) ? (s >> 5) : 5;
      int kt = (s < 160) ? (s & 31) : (s - 160);
      const uint4* src = (const uint4*)(wgsrc + (size_t)(g * 32 + kt) * KT_STRIDE + (size_t)cc * 8);
      *(uint4*)(sW + (size_t)s * TILE_E + (size_t)cc * 8) = *src;
    }
  }

  // loop-invariant per-lane constants
  const float b_i = bih[0 * HH + n] + bhh[0 * HH + n];
  const float b_f = bih[1 * HH + n] + bhh[1 * HH + n];
  const float b_g = bih[2 * HH + n] + bhh[2 * HH + n];
  const float b_o = bih[3 * HH + n] + bhh[3 * HH + n];
  const float b_r = bgi[n] + bgs[n];
  const int   tr  = *train;

  float dmv[8], hprev[8], cprev[8];
#pragma unroll
  for (int j = 0; j < 8; ++j) {
    int bat = wave * 16 + mrow + j;
    dmv[j] = tr ? dmask[(size_t)bat * HH + n] : 1.0f;
    hprev[j] = 0.0f;
    cprev[j] = 0.0f;
  }
  __syncthreads();   // weight slice resident

  const __bf16* sA_w = sA + (size_t)(wave * 32) * TILE_E + (size_t)ln * 16;
  const __bf16* sW_l = sW + (size_t)ln * 16;

  for (int t = 0; t < SQ; ++t) {
    float* hc = hbuf + (size_t)(t & 1) * (BB * HH);         // read (prev h, all columns)
    float* hx = hbuf + (size_t)((t + 1) & 1) * (BB * HH);   // write (new h)

    // prefetch next timestep's input row while we work on this one
    if (t + 1 < SQ)
      __builtin_prefetch((const void*)(xin + ((size_t)(t + 1) * BB) * HH), 0, 1);

    // ---- pack A = [x_t ; h_prev] into LDS bf16 A-tiles ----
    for (int u = threadIdx.x; u < 2048; u += 64) {
      int tile  = u >> 5;            // 0..63
      int l2    = u & 31;            // lane-slot inside tile
      int mtile = tile >> 5;
      int kt    = tile & 31;
      int bat   = mtile * 16 + (l2 & 15);
      int kbase = kt * 32 + 16 * (l2 >> 4);
      __bf16* dst = &sA[(size_t)tile * TILE_E + (size_t)l2 * 16];
#pragma unroll
      for (int h8 = 0; h8 < 2; ++h8) {
        int k0 = kbase + h8 * 8;
        const float* src = (k0 < HH)
            ? (xin + ((size_t)t * BB + bat) * HH + k0)
            : (hc + (size_t)bat * HH + (k0 - HH));
#pragma unroll
        for (int q = 0; q < 8; ++q) dst[h8 * 8 + q] = f2bf(src[q]);
      }
    }
    __syncthreads();

    // ---- GEMM from LDS: 6 accumulator tiles, K = 1024 (x+h); lin group only K = 512 ----
    v8f acc0 = {}, acc1 = {}, acc2 = {}, acc3 = {}, acc4 = {}, acc5 = {};
#pragma unroll 2
    for (int kt = 0; kt < 16; ++kt) {                 // x half of K: all 6 groups
      // hoist all operand loads into distinct vectors -> one ds_load clause per K-tile
      const v16bf a  = *(const v16bf*)(sA_w + (size_t)kt * TILE_E);
      const v16bf w0 = *(const v16bf*)(sW_l + (size_t)(0 * 32 + kt) * TILE_E);
      const v16bf w1 = *(const v16bf*)(sW_l + (size_t)(1 * 32 + kt) * TILE_E);
      const v16bf w2 = *(const v16bf*)(sW_l + (size_t)(2 * 32 + kt) * TILE_E);
      const v16bf w3 = *(const v16bf*)(sW_l + (size_t)(3 * 32 + kt) * TILE_E);
      const v16bf w4 = *(const v16bf*)(sW_l + (size_t)(4 * 32 + kt) * TILE_E);
      const v16bf w5 = *(const v16bf*)(sW_l + (size_t)(160 + kt) * TILE_E);
      acc0 = WMMA_BF16(a, w0, acc0);
      acc1 = WMMA_BF16(a, w1, acc1);
      acc2 = WMMA_BF16(a, w2, acc2);
      acc3 = WMMA_BF16(a, w3, acc3);
      acc4 = WMMA_BF16(a, w4, acc4);
      acc5 = WMMA_BF16(a, w5, acc5);
    }
#pragma unroll 2
    for (int kt = 16; kt < 32; ++kt) {                // h half of K: groups 0..4
      const v16bf a  = *(const v16bf*)(sA_w + (size_t)kt * TILE_E);
      const v16bf w0 = *(const v16bf*)(sW_l + (size_t)(0 * 32 + kt) * TILE_E);
      const v16bf w1 = *(const v16bf*)(sW_l + (size_t)(1 * 32 + kt) * TILE_E);
      const v16bf w2 = *(const v16bf*)(sW_l + (size_t)(2 * 32 + kt) * TILE_E);
      const v16bf w3 = *(const v16bf*)(sW_l + (size_t)(3 * 32 + kt) * TILE_E);
      const v16bf w4 = *(const v16bf*)(sW_l + (size_t)(4 * 32 + kt) * TILE_E);
      acc0 = WMMA_BF16(a, w0, acc0);
      acc1 = WMMA_BF16(a, w1, acc1);
      acc2 = WMMA_BF16(a, w2, acc2);
      acc3 = WMMA_BF16(a, w3, acc3);
      acc4 = WMMA_BF16(a, w4, acc4);
    }

    // ---- elementwise LSTM cell + r-gate highway + dropout + sequence mask ----
    const int tm = (layer == 0) ? t : (SQ - 1 - t);   // mask is time-reversed for layer 1
#pragma unroll
    for (int j = 0; j < 8; ++j) {
      int   bat = wave * 16 + mrow + j;
      float mt  = mask[(size_t)tm * BB + bat];
      float iv  = sigf(acc0[j] + b_i);
      float fv  = sigf(acc1[j] + b_f);
      float gv  = tanhf_(acc2[j] + b_g);
      float ov  = sigf(acc3[j] + b_o);
      float rv  = sigf(acc4[j] + b_r);
      float lv  = acc5[j];
      float ct  = fv * cprev[j] + iv * gv;
      float ht  = ov * tanhf_(ct);
      ht = rv * ht + (1.0f - rv) * lv;
      ht *= dmv[j];
      float hn = mt * ht + (1.0f - mt) * hprev[j];
      float cn = mt * ct + (1.0f - mt) * cprev[j];
      hprev[j] = hn;
      cprev[j] = cn;
      hx[(size_t)bat * HH + n] = hn;                  // feed next step's GEMM
      if (layer == 0)                                 // time-reversed activation for layer 1
        outp[((size_t)(SQ - 1 - t) * BB + bat) * HH + n] = hn;
      else                                            // final (B, S, H) with time reversal
        outp[((size_t)bat * SQ + (SQ - 1 - t)) * HH + n] = hn;
    }

    // ---- device-wide step barrier (all 32 WGs resident) ----
    __syncthreads();
    if (threadIdx.x == 0) {
      __threadfence();
      unsigned old = atomicAdd(&sync[0], 1u);
      if (old == NWG - 1u) {
        atomicExch(&sync[0], 0u);
        __threadfence();
        atomicAdd(&sync[1], 1u);
      } else {
        while (__hip_atomic_load(&sync[1], __ATOMIC_ACQUIRE, __HIP_MEMORY_SCOPE_AGENT) < (unsigned)(t + 1))
          __builtin_amdgcn_s_sleep(1);
      }
    }
    __syncthreads();
  }
}

// ---------------- host launcher ----------------
extern "C" void kernel_launch(void* const* d_in, const int* in_sizes, int n_in,
                              void* d_out, int out_size, void* d_ws, size_t ws_size,
                              hipStream_t stream) {
  const float* x     = (const float*)d_in[0];
  const float* mask  = (const float*)d_in[1];
  const float* Wih   = (const float*)d_in[2];
  const float* bih   = (const float*)d_in[3];
  const float* Whh   = (const float*)d_in[4];
  const float* bhh   = (const float*)d_in[5];
  const float* Wgi   = (const float*)d_in[6];
  const float* bgi   = (const float*)d_in[7];
  const float* Wgs   = (const float*)d_in[8];
  const float* bgs   = (const float*)d_in[9];
  const float* Wli   = (const float*)d_in[10];
  const float* dmask = (const float*)d_in[11];
  const int*   train = (const int*)d_in[12];

  // workspace layout (~46.3 MB): sync | h ping-pong | packed bf16 weights | inter-layer activations
  char*     ws    = (char*)d_ws;
  unsigned* sync  = (unsigned*)(ws + 0);
  float*    hbuf  = (float*)(ws + 4096);                                // 2*32*512 f32 = 128 KB
  __bf16*   wpack = (__bf16*)(ws + 4096 + 2 * BB * HH * sizeof(float)); // 12 MB bf16
  float*    X1    = (float*)(ws + 4096 + 2 * BB * HH * sizeof(float) + (size_t)NPACK * 2); // 32 MB

  pack_w_kernel<<<(NPACK + 255) / 256, 256, 0, stream>>>(Wih, Whh, Wgi, Wgs, Wli, wpack);

  for (int layer = 0; layer < 2; ++layer) {
    init_kernel<<<(2 * BB * HH + 255) / 256, 256, 0, stream>>>(sync, hbuf);
    recur_kernel<<<NWG, 64, SMEM_BYTES, stream>>>(
        layer ? X1 : x, mask,
        bih + (size_t)layer * 4 * HH, bhh + (size_t)layer * 4 * HH,
        bgi + (size_t)layer * HH, bgs,
        dmask + (size_t)layer * BB * HH, train,
        wpack + (size_t)layer * LAYER_E, hbuf, sync,
        layer ? (float*)d_out : X1, layer);
  }
}
